// Decoder_29987461660830
// MI455X (gfx1250) — matrix operand order repaired and checked by reading, not compiled, not run on hardware
//
#include <hip/hip_runtime.h>
#include <cstdint>
#include <cstddef>
#include <type_traits>

typedef __bf16 bf16;
typedef __attribute__((ext_vector_type(16))) __bf16 v16bf;
typedef __attribute__((ext_vector_type(8)))  __bf16 v8bf;
typedef __attribute__((ext_vector_type(8)))  float  v8f;
typedef __attribute__((ext_vector_type(4)))  unsigned int v4u;
typedef __attribute__((ext_vector_type(8)))  int v8i_t;
typedef __attribute__((ext_vector_type(4)))  int v4i_t;

#define DEV __device__ __forceinline__

// Detect the TDM builtin (device pass only) and its arity flavor.
#if defined(__AMDGCN__) && __has_builtin(__builtin_amdgcn_tensor_load_to_lds)
#define USE_TDM 1
#else
#define USE_TDM 0
#endif
#if USE_TDM && __has_include(<hip/amd_detail/amd_gfx1250_TDM.h>)
#define TDM_ARGS6 1
#else
#define TDM_ARGS6 0
#endif

// Problem sizes (fixed by the reference)
constexpr int NB  = 128;    // batch
constexpr int TT  = 512;    // encoder steps
constexpr int HE2 = 1024;   // 2*encoder_hidden
constexpr int EE  = 256;    // embedding size
constexpr int DKV = 256;    // key/value size
constexpr int DHH = 1024;   // decoder hidden
constexpr int VV  = 4096;   // vocab
constexpr int LL  = 200;    // decode steps
constexpr int LP  = 208;    // L padded to multiple of 16

union VU { v16bf v; v8bf h[2]; };

// A-fragment (16x32 bf16, MxK): lane<16 row=lane, k in [half*16,+16)
DEV v16bf load_a_frag(const bf16* base, int stride) {
  int lane = threadIdx.x & 31;
  const bf16* p = base + (lane & 15) * stride + ((lane >> 4) * 16);
  VU u;
  u.h[0] = *(const v8bf*)p;
  u.h[1] = *(const v8bf*)(p + 8);
  return u.v;
}
// B-fragment (32x16 bf16, KxN): lane holds one column, 16 contiguous k's
DEV v16bf load_b_frag(const bf16* base, int stride) {
  int lane = threadIdx.x & 31;
  const bf16* p = base + (lane & 15) * stride + ((lane >> 4) * 16);
  VU u;
  u.h[0] = *(const v8bf*)p;
  u.h[1] = *(const v8bf*)(p + 8);
  return u.v;
}
DEV v8f wmma_bf16(v16bf a, v16bf b, v8f c) {
  return __builtin_amdgcn_wmma_f32_16x16x32_bf16(false, a, false, b, (short)0, c,
                                                 false, false);
}
DEV bf16 to_bf(float x) { return (bf16)x; }
DEV bf16 to_bf(bf16 x)  { return x; }

// ---------------------------------------------------------------------------
// TDM: DMA a (tile_rows x 64) bf16 tile into LDS, padding 4 DWORDs every
// 32 DWORDs so the LDS row stride is 72 elements (matches fragment loads).
// D# group0/group1 packed per CDNA5 ISA 8.3/8.4.  One issue per wave 0.
// ---------------------------------------------------------------------------
#if USE_TDM
DEV void tdm_load_tile(unsigned int lds_off, const bf16* gaddr, int tile_rows,
                       int tdim0, int tdim1, int stride_elems) {
  unsigned long long ga = (unsigned long long)(uintptr_t)gaddr;
  v4u g0;
  g0.x = 1u;                                   // count=1, user descriptor
  g0.y = lds_off;                              // LDS byte address
  g0.z = (unsigned int)ga;                     // global_addr[31:0]
  g0.w = (unsigned int)((ga >> 32) & 0x1FFFFFFull) | (2u << 30);  // type=2
  v8i_t g1;
  // data_size=1 (2B) | pad_enable | pad_interval=4 (32 DW) | pad_amount=3 (4 DW)
  g1[0] = (int)((1u << 16) | (1u << 20) | (4u << 22) | (3u << 25));
  g1[1] = (int)(((unsigned)tdim0 & 0xFFFFu) << 16);                // dim0[15:0]
  g1[2] = (int)((((unsigned)tdim0 >> 16) & 0xFFFFu) |
                (((unsigned)tdim1 & 0xFFFFu) << 16));              // dim0 hi | dim1 lo
  g1[3] = (int)((((unsigned)tdim1 >> 16) & 0xFFFFu) | (64u << 16)); // dim1 hi | tile_dim0=64
  g1[4] = (int)((unsigned)tile_rows & 0xFFFFu);                    // tile_dim1
  g1[5] = stride_elems;                                            // dim0_stride[31:0]
  g1[6] = 0;
  g1[7] = 0;
  v4i_t z4 = {0, 0, 0, 0};
#if TDM_ARGS6
  v8i_t z8 = {0, 0, 0, 0, 0, 0, 0, 0};
  __builtin_amdgcn_tensor_load_to_lds(g0, g1, z4, z4, z8, 0);
#else
  __builtin_amdgcn_tensor_load_to_lds(g0, g1, z4, z4, 0);
#endif
}
#else
DEV void tdm_load_tile(unsigned int, const bf16*, int, int, int, int) {}
#endif

// ---------------------------------------------------------------------------
// Double-buffered WMMA GEMM: C[M,Nout] = A[M,K] * Bw[Nout,K]^T (+bias)(+Cadd)
// Tile 128(M) x 64(N) x 64(K); 8 waves in 4(M)x2(N); wave tile 32x32 (4 accs).
// B tiles (and A tiles when already bf16) staged by the Tensor Data Mover;
// f32 A tiles staged with vector loads + convert.  TENSORcnt guards publish.
// remapLp>0: output row (n*remapLp + l) -> (n*remapL + l), skip l>=remapL.
// ---------------------------------------------------------------------------
template<typename TA, typename TC>
__global__ __launch_bounds__(256)
void gemm_bf16_wmma(const TA* __restrict__ A, int lda,
                    const bf16* __restrict__ Bw, int ldb,
                    const float* __restrict__ bias,
                    const float* __restrict__ Cadd, size_t ldadd,
                    TC* __restrict__ C, int ldc,
                    int M, int Nout, int K,
                    int remapLp, int remapL)
{
  constexpr int BM = 128, BN = 64, BK = 64;
  constexpr int SAB = 72;   // LDS row stride in bf16 (144B, 16B-aligned)
  __shared__ bf16 a_s[2][BM * SAB];
  __shared__ bf16 b_s[2][BN * SAB];

  const int tid  = threadIdx.x;
  const int m0   = blockIdx.y * BM;
  const int n0   = blockIdx.x * BN;
  const int wave = tid >> 5;
  const int wm   = (wave >> 1) * 32;
  const int wn   = (wave & 1) * 32;

  constexpr bool A_TDM = (USE_TDM != 0) && std::is_same<TA, bf16>::value;

  auto stage_a = [&](int buf, int kk0) {
    if constexpr (A_TDM) {
      if (wave == 0)
        tdm_load_tile((unsigned int)(uintptr_t)&a_s[buf][0],
                      reinterpret_cast<const bf16*>(A) + (size_t)m0 * lda + kk0,
                      BM, K - kk0, M - m0, lda);
    } else {
      for (int i = tid; i < BM * BK; i += 256) {
        int r = i >> 6, k = i & 63;
        int row = m0 + r, kg = kk0 + k;
        bf16 vv = (bf16)0.0f;
        if (row < M && kg < K) vv = to_bf(A[(size_t)row * lda + kg]);
        a_s[buf][r * SAB + k] = vv;
      }
    }
  };
  auto stage_b = [&](int buf, int kk0) {
    if constexpr (USE_TDM != 0) {
      if (wave == 0)
        tdm_load_tile((unsigned int)(uintptr_t)&b_s[buf][0],
                      Bw + (size_t)n0 * ldb + kk0, BN, K - kk0, Nout - n0, ldb);
    } else {
      for (int i = tid; i < BN * BK; i += 256) {
        int r = i >> 6, k = i & 63;
        int col = n0 + r, kg = kk0 + k;
        bf16 vv = (bf16)0.0f;
        if (col < Nout && kg < K) vv = Bw[(size_t)col * ldb + kg];
        b_s[buf][r * SAB + k] = vv;
      }
    }
  };

  v8f acc[2][2] = {};

  const int nstage = (K + BK - 1) / BK;
  stage_b(0, 0);
  stage_a(0, 0);
#if USE_TDM
  if (wave == 0) __builtin_amdgcn_s_wait_tensorcnt((short)0);
#endif
  __syncthreads();

  for (int it = 0; it < nstage; ++it) {
    const int cur = it & 1;
    const int kk0 = it * BK;
    if (it + 1 < nstage) {      // overlap next-tile staging with WMMA
      stage_b(cur ^ 1, kk0 + BK);
      stage_a(cur ^ 1, kk0 + BK);
    }
    #pragma unroll
    for (int ks = 0; ks < BK; ks += 32) {
      v16bf af0 = load_a_frag(&a_s[cur][(wm + 0)  * SAB + ks], SAB);
      v16bf af1 = load_a_frag(&a_s[cur][(wm + 16) * SAB + ks], SAB);
      v16bf bf0 = load_b_frag(&b_s[cur][(wn + 0)  * SAB + ks], SAB);
      v16bf bf1 = load_b_frag(&b_s[cur][(wn + 16) * SAB + ks], SAB);
      acc[0][0] = wmma_bf16(af0, bf0, acc[0][0]);
      acc[0][1] = wmma_bf16(af0, bf1, acc[0][1]);
      acc[1][0] = wmma_bf16(af1, bf0, acc[1][0]);
      acc[1][1] = wmma_bf16(af1, bf1, acc[1][1]);
    }
#if USE_TDM
    if (wave == 0 && it + 1 < nstage)
      __builtin_amdgcn_s_wait_tensorcnt((short)0);
#endif
    __syncthreads();
  }

  // Store: C/D layout => lane<16: (M=v, N=lane); lane>=16: (M=v+8, N=lane-16)
  const int lane = tid & 31;
  const int half = lane >> 4;
  const int nl   = lane & 15;
  for (int mi = 0; mi < 2; ++mi) {
    for (int nj = 0; nj < 2; ++nj) {
      int col = n0 + wn + nj * 16 + nl;
      if (col >= Nout) continue;
      float bb = bias ? bias[col] : 0.0f;
      #pragma unroll
      for (int v = 0; v < 8; ++v) {
        int row = m0 + wm + mi * 16 + half * 8 + v;
        if (row >= M) continue;
        float val = acc[mi][nj][v] + bb;
        if (Cadd) val += Cadd[(size_t)row * ldadd + col];
        size_t orow = (size_t)row;
        if (remapLp > 0) {
          int nn = row / remapLp;
          int ll = row - nn * remapLp;
          if (ll >= remapL) continue;
          orow = (size_t)nn * remapL + ll;
        }
        C[orow * (size_t)ldc + col] = (TC)val;
      }
    }
  }
}

// ---------------------------------------------------------------------------
// Fused attention: one block per (batch n, 16-row L-tile).
//   E[16,512] = CE[16,256] x key[n]^T   (WMMA)
//   masked softmax rows (len = lens[n]); probs unnormalized in bf16
//   CTX[16,256] = P[16,512] x value[n]  (WMMA), scaled by 1/sum at store
// ctx written into x[:, 256:512] (x = [ce || ctx] feeds the W_ih GEMM).
// ---------------------------------------------------------------------------
__global__ __launch_bounds__(256)
void attention_kernel(bf16* __restrict__ x,            // [NB*LP, 512]
                      const bf16* __restrict__ keyb,   // [NB*TT, 256]
                      const bf16* __restrict__ valb,   // [NB*TT, 256]
                      const int* __restrict__ lens)
{
  __shared__ bf16  a_s[16 * 264];
  __shared__ float e_s[16 * 512];
  __shared__ bf16  p_s[16 * 512];
  __shared__ float red_s[16 * 16];
  __shared__ float rowmax_s[16];
  __shared__ float inv_s[16];

  const int tid  = threadIdx.x;
  const int n    = blockIdx.y;
  const int l0   = blockIdx.x * 16;
  const int wave = tid >> 5;
  const int lane = tid & 31;
  const int half = lane >> 4;
  const int nl   = lane & 15;

  for (int i = tid; i < 16 * 256; i += 256) {
    int r = i >> 8, k = i & 255;
    a_s[r * 264 + k] = x[((size_t)(n * LP + l0 + r)) * 512 + k];
  }
  __syncthreads();

  {
    const int t0 = wave * 64;
    v8f acc[4] = {};
    for (int k0 = 0; k0 < 256; k0 += 32) {
      v16bf af = load_a_frag(&a_s[k0], 264);
      #pragma unroll
      for (int j = 0; j < 4; ++j) {
        v16bf bfj = load_b_frag(
            &keyb[((size_t)(n * TT + t0 + j * 16)) * 256 + k0], 256);
        acc[j] = wmma_bf16(af, bfj, acc[j]);
      }
    }
    #pragma unroll
    for (int j = 0; j < 4; ++j)
      #pragma unroll
      for (int v = 0; v < 8; ++v)
        e_s[(half * 8 + v) * 512 + t0 + j * 16 + nl] = acc[j][v];
  }
  __syncthreads();

  const int Tlen = lens[n];
  {
    int row = tid >> 4, sub = tid & 15;
    float mx = -3.4e38f;
    for (int t = sub; t < TT; t += 16)
      if (t < Tlen) mx = fmaxf(mx, e_s[row * 512 + t]);
    red_s[row * 16 + sub] = mx;
    __syncthreads();
    if (sub == 0) {
      float m2 = -3.4e38f;
      for (int j2 = 0; j2 < 16; ++j2) m2 = fmaxf(m2, red_s[row * 16 + j2]);
      rowmax_s[row] = m2;
    }
    __syncthreads();
    float rm = rowmax_s[row];
    float sm = 0.0f;
    for (int t = sub; t < TT; t += 16) {
      float pv = 0.0f;
      if (t < Tlen) pv = __expf(e_s[row * 512 + t] - rm);
      p_s[row * 512 + t] = (bf16)pv;
      sm += pv;
    }
    red_s[row * 16 + sub] = sm;
    __syncthreads();
    if (sub == 0) {
      float s2 = 0.0f;
      for (int j2 = 0; j2 < 16; ++j2) s2 += red_s[row * 16 + j2];
      inv_s[row] = 1.0f / s2;
    }
  }
  __syncthreads();

  {
    const int d0 = wave * 32;
    v8f acc[2] = {};
    for (int k0 = 0; k0 < TT; k0 += 32) {
      v16bf af = load_a_frag(&p_s[k0], 512);
      #pragma unroll
      for (int j = 0; j < 2; ++j) {
        const bf16* vp =
            &valb[((size_t)(n * TT + k0 + half * 16)) * 256 + d0 + j * 16 + nl];
        VU u;
        #pragma unroll
        for (int e = 0; e < 8; ++e) u.h[0][e] = vp[(size_t)e * 256];
        #pragma unroll
        for (int e = 0; e < 8; ++e) u.h[1][e] = vp[(size_t)(8 + e) * 256];
        acc[j] = wmma_bf16(af, u.v, acc[j]);
      }
    }
    #pragma unroll
    for (int j = 0; j < 2; ++j)
      #pragma unroll
      for (int v = 0; v < 8; ++v) {
        int m = half * 8 + v;
        float sc = inv_s[m];
        x[((size_t)(n * LP + l0 + m)) * 512 + 256 + d0 + j * 16 + nl] =
            (bf16)(acc[j][v] * sc);
      }
  }
}

// ---------------------------------------------------------------------------
// Small helper kernels
// ---------------------------------------------------------------------------
__global__ void f2bf_kernel(const float* __restrict__ s, bf16* __restrict__ d,
                            int n) {
  int i = blockIdx.x * 256 + threadIdx.x;
  if (i < n) d[i] = (bf16)s[i];
}

__global__ void bias_sum_kernel(const float* __restrict__ a,
                                const float* __restrict__ b,
                                float* __restrict__ o, int n) {
  int i = blockIdx.x * 256 + threadIdx.x;
  if (i < n) o[i] = a[i] + b[i];
}

__global__ void init_state_kernel(float* __restrict__ c, bf16* __restrict__ h,
                                  int n) {
  int i = blockIdx.x * 256 + threadIdx.x;
  if (i < n) { c[i] = 0.0f; h[i] = (bf16)0.0f; }
}

__global__ void embed_kernel(const int* __restrict__ text,
                             const float* __restrict__ emb,
                             bf16* __restrict__ x) {
  int i = blockIdx.x * 256 + threadIdx.x;       // over NB*LP*EE
  if (i >= NB * LP * EE) return;
  int d = i & 255;
  int r = i >> 8;                                // n*LP + l
  int n = r / LP, l = r - n * LP;
  int tok = 0;
  if (l > 0 && l < LL) tok = text[n * LL + l - 1];
  x[(size_t)r * 512 + d] = (bf16)emb[(size_t)tok * EE + d];
}

__global__ void lstm_cell_kernel(const float* __restrict__ gates,
                                 float* __restrict__ c,
                                 bf16* __restrict__ hprev,
                                 bf16* __restrict__ h_all, int l) {
  int i = blockIdx.x * 256 + threadIdx.x;        // over NB*DHH
  if (i >= NB * DHH) return;
  int n = i >> 10, j = i & 1023;
  size_t gb = (size_t)n * 4096;
  float ig = gates[gb + j];
  float fg = gates[gb + 1024 + j];
  float gg = gates[gb + 2048 + j];
  float og = gates[gb + 3072 + j];
  float is = 1.0f / (1.0f + __expf(-ig));
  float fs = 1.0f / (1.0f + __expf(-fg));
  float gt = tanhf(gg);
  float os = 1.0f / (1.0f + __expf(-og));
  float cn = fs * c[i] + is * gt;
  c[i] = cn;
  bf16 hb = (bf16)(os * tanhf(cn));
  hprev[i] = hb;
  h_all[((size_t)n * LP + l) * 1024 + j] = hb;
}

// ---------------------------------------------------------------------------
extern "C" void kernel_launch(void* const* d_in, const int* in_sizes, int n_in,
                              void* d_out, int out_size, void* d_ws,
                              size_t ws_size, hipStream_t stream) {
  (void)in_sizes; (void)n_in; (void)out_size; (void)ws_size;

  const float* encoder_out = (const float*)d_in[0];
  const int*   text  = (const int*)d_in[1];
  const int*   lens  = (const int*)d_in[2];
  const float* emb   = (const float*)d_in[3];
  const float* W_key = (const float*)d_in[4];
  const float* b_key = (const float*)d_in[5];
  const float* W_val = (const float*)d_in[6];
  const float* b_val = (const float*)d_in[7];
  const float* W_ih  = (const float*)d_in[8];
  const float* b_ih  = (const float*)d_in[9];
  const float* W_hh  = (const float*)d_in[10];
  const float* b_hh  = (const float*)d_in[11];
  const float* W_out = (const float*)d_in[12];
  const float* b_out = (const float*)d_in[13];
  float* preds = (float*)d_out;

  char* wp = (char*)d_ws;
  auto alloc = [&](size_t bytes) -> void* {
    void* r = (void*)wp;
    wp += (bytes + 255) & ~(size_t)255;
    return r;
  };

  bf16*  Wkey_b  = (bf16*)alloc((size_t)DKV * HE2 * 2);
  bf16*  Wval_b  = (bf16*)alloc((size_t)DKV * HE2 * 2);
  bf16*  Wih_b   = (bf16*)alloc((size_t)4 * DHH * 512 * 2);
  bf16*  Whh_b   = (bf16*)alloc((size_t)4 * DHH * DHH * 2);
  bf16*  Wout_b  = (bf16*)alloc((size_t)VV * DHH * 2);
  float* b_sum   = (float*)alloc((size_t)4 * DHH * 4);
  bf16*  key_b   = (bf16*)alloc((size_t)NB * TT * DKV * 2);
  bf16*  val_b   = (bf16*)alloc((size_t)NB * TT * DKV * 2);
  bf16*  x       = (bf16*)alloc((size_t)NB * LP * 512 * 2);   // [ce || ctx]
  float* gates_x = (float*)alloc((size_t)NB * LP * 4096 * 4);
  bf16*  h_all   = (bf16*)alloc((size_t)NB * LP * DHH * 2);
  bf16*  hprev   = (bf16*)alloc((size_t)NB * DHH * 2);
  float* c_st    = (float*)alloc((size_t)NB * DHH * 4);
  float* gates_st= (float*)alloc((size_t)NB * 4096 * 4);

  auto cvt = [&](const float* s, bf16* d, int n) {
    f2bf_kernel<<<(n + 255) / 256, 256, 0, stream>>>(s, d, n);
  };
  cvt(W_key, Wkey_b, DKV * HE2);
  cvt(W_val, Wval_b, DKV * HE2);
  cvt(W_ih,  Wih_b,  4 * DHH * 512);
  cvt(W_hh,  Whh_b,  4 * DHH * DHH);
  cvt(W_out, Wout_b, VV * DHH);
  bias_sum_kernel<<<16, 256, 0, stream>>>(b_ih, b_hh, b_sum, 4 * DHH);
  init_state_kernel<<<(NB * DHH + 255) / 256, 256, 0, stream>>>(c_st, hprev,
                                                                NB * DHH);
  embed_kernel<<<(NB * LP * EE + 255) / 256, 256, 0, stream>>>(text, emb, x);

  // Encoder projections: key/value = encoder_out @ W^T + b   [N*T, 256]
  {
    dim3 g(DKV / 64, (NB * TT) / 128);
    gemm_bf16_wmma<float, bf16><<<g, 256, 0, stream>>>(
        encoder_out, HE2, Wkey_b, HE2, b_key, nullptr, 0,
        key_b, DKV, NB * TT, DKV, HE2, 0, 0);
    gemm_bf16_wmma<float, bf16><<<g, 256, 0, stream>>>(
        encoder_out, HE2, Wval_b, HE2, b_val, nullptr, 0,
        val_b, DKV, NB * TT, DKV, HE2, 0, 0);
  }

  // Attention for all decode steps at once
  attention_kernel<<<dim3(LP / 16, NB), 256, 0, stream>>>(x, key_b, val_b,
                                                          lens);

  // gates_x = [ce||ctx] @ W_ih^T + (b_ih + b_hh)   [N*LP, 4096]
  gemm_bf16_wmma<bf16, float><<<dim3(4 * DHH / 64, NB * LP / 128), 256, 0,
                                stream>>>(
      x, 512, Wih_b, 512, b_sum, nullptr, 0,
      gates_x, 4 * DHH, NB * LP, 4 * DHH, 512, 0, 0);

  // Sequential LSTM recurrence
  for (int l = 0; l < LL; ++l) {
    gemm_bf16_wmma<bf16, float><<<dim3(4 * DHH / 64, 1), 256, 0, stream>>>(
        hprev, DHH, Whh_b, DHH, nullptr,
        gates_x + (size_t)l * 4096, (size_t)LP * 4096,
        gates_st, 4 * DHH, NB, 4 * DHH, DHH, 0, 0);
    lstm_cell_kernel<<<(NB * DHH + 255) / 256, 256, 0, stream>>>(
        gates_st, c_st, hprev, h_all, l);
  }

  // preds = H @ W_out^T + b_out, dropping the L-padding rows on store
  gemm_bf16_wmma<bf16, float><<<dim3(VV / 64, NB * LP / 128), 256, 0,
                                stream>>>(
      h_all, DHH, Wout_b, DHH, b_out, nullptr, 0,
      preds, VV, NB * LP, VV, DHH, LP, LL);
}
